// SoftMixtureOfMixers_75007308857797
// MI455X (gfx1250) — compile-verified, matrix operand order checked by reading, not run on hardware
//
#include <hip/hip_runtime.h>
#include <math.h>

// ---------------------------------------------------------------------------
// Soft Mixture-of-Mixers for MI455X (gfx1250, wave32, WMMA bf16 16x16x32)
//
// Pipeline (all GEMMs NT-form:  C[m,n] = sum_k A[m,k] * W[n,k]):
//  1. transpose/convert:    x(f32) -> x_bf (B,N,D) and xT_bf (B,D,N)
//  2. convert phi_w, out_w -> bf16
//  3. logits  = x_bf  @ phi_bf^T                 (32768 x 128  x 1024)
//  4. combine softmax over T (rows)  -> comb_bf   (B,N,T)
//     dispatch softmax over N (cols) -> dispT_bf  (B,T,N)
//  5. slots   = dispT @ xT^T  per batch          (128 x 1024 x 4096)
//  6. expert MLP (per e): h = gelu(slots_e @ fc1^T + b1)     f32 weights,
//     out = h @ fc2^T + b2 -> written transposed outT (B,D,T)
//  7. combined = comb @ outT^T per batch         (4096 x 1024 x 128)
//  8. output   = combined @ out_w^T + out_b      (32768 x 1024 x 1024) -> f32
//  9. aux_loss = 0
// ---------------------------------------------------------------------------

typedef __attribute__((ext_vector_type(16))) __bf16 v16bf;
typedef __attribute__((ext_vector_type(8)))  __bf16 v8bf;
typedef __attribute__((ext_vector_type(8)))  float  v8f;
typedef __attribute__((ext_vector_type(4)))  float  v4f;

union FragBF { v16bf v; v8bf h[2]; __bf16 e[16]; };

__device__ __forceinline__ float gelu_tanh(float x) {
    const float c = 0.7978845608028654f;   // sqrt(2/pi)
    return 0.5f * x * (1.0f + tanhf(c * (x + 0.044715f * x * x * x)));
}

// ---------------------------------------------------------------------------
// Generic NT GEMM, bf16 inputs, f32 accumulate.
// Block = 256 threads (8 waves). Tile 128(M) x 128(N) x 32(K).
// Waves: 4 along M (32 rows each), 2 along N (64 cols each); 2x4 WMMA tiles.
// Double-buffered LDS: one barrier per K-step; next slab is fetched to
// registers during compute and stored into the alternate buffer.
// M, N, K are assumed multiples of 128/128/32 (true for every call site).
// ---------------------------------------------------------------------------
template<bool OUT_BF16, bool HAS_BIAS>
__global__ __launch_bounds__(256)
void gemm_nt_kernel(const __bf16* __restrict__ Aall, long long sAb,
                    const __bf16* __restrict__ Wall, long long sWb,
                    const float*  __restrict__ bias,
                    void* __restrict__ Call, long long sCb,
                    int K, int lda, int ldw, int ldc)
{
    const int bz  = blockIdx.z;
    const __bf16* A = Aall + (long long)bz * sAb;
    const __bf16* W = Wall + (long long)bz * sWb;
    const int gm0 = blockIdx.y * 128;
    const int gn0 = blockIdx.x * 128;

    // padded rows: 32 data + 8 pad bf16  (80 bytes/row -> conflict-free b128 reads)
    __shared__ __align__(16) __bf16 sA[2][128 * 40];
    __shared__ __align__(16) __bf16 sB[2][128 * 40];

    const int tid  = threadIdx.x;
    const int lane = tid & 31;
    const int wid  = tid >> 5;
    const int wm   = (wid & 3) * 32;     // wave M offset
    const int wn   = (wid >> 2) * 64;    // wave N offset
    const int lm   = lane & 15;
    const int hh   = lane >> 4;

    const int ar = tid >> 2;             // 0..63  (row within tile, two passes)
    const int ac = (tid & 3) * 8;        // 0,8,16,24

    v8f acc[2][4] = {};
    v8bf ra[2], rb[2];                   // register-staged next slab

    auto gload = [&](int k0) {
        #pragma unroll
        for (int p = 0; p < 2; ++p) {
            const int row = ar + p * 64;
            ra[p] = *reinterpret_cast<const v8bf*>(&A[(long long)(gm0 + row) * lda + k0 + ac]);
            rb[p] = *reinterpret_cast<const v8bf*>(&W[(long long)(gn0 + row) * ldw + k0 + ac]);
        }
    };
    auto lstore = [&](int buf) {
        #pragma unroll
        for (int p = 0; p < 2; ++p) {
            const int row = ar + p * 64;
            *reinterpret_cast<v8bf*>(&sA[buf][row * 40 + ac]) = ra[p];
            *reinterpret_cast<v8bf*>(&sB[buf][row * 40 + ac]) = rb[p];
        }
    };

    gload(0);
    lstore(0);
    int cur = 0;

    for (int k0 = 0; k0 < K; k0 += 32) {
        __syncthreads();                       // buf[cur] published; buf[cur^1] free
        const bool hasNext = (k0 + 32) < K;
        if (hasNext) gload(k0 + 32);

        FragBF af[2], bfr[4];
        #pragma unroll
        for (int mi = 0; mi < 2; ++mi) {
            const int m = wm + mi * 16 + lm;
            af[mi].h[0] = *reinterpret_cast<const v8bf*>(&sA[cur][m * 40 + hh * 8]);
            af[mi].h[1] = *reinterpret_cast<const v8bf*>(&sA[cur][m * 40 + 16 + hh * 8]);
        }
        #pragma unroll
        for (int ni = 0; ni < 4; ++ni) {
            const int n = wn + ni * 16 + lm;
            bfr[ni].h[0] = *reinterpret_cast<const v8bf*>(&sB[cur][n * 40 + hh * 8]);
            bfr[ni].h[1] = *reinterpret_cast<const v8bf*>(&sB[cur][n * 40 + 16 + hh * 8]);
        }
        #pragma unroll
        for (int mi = 0; mi < 2; ++mi)
            #pragma unroll
            for (int ni = 0; ni < 4; ++ni)
                acc[mi][ni] = __builtin_amdgcn_wmma_f32_16x16x32_bf16(
                    false, af[mi].v, false, bfr[ni].v,
                    (short)0, acc[mi][ni], false, false);

        if (hasNext) lstore(cur ^ 1);
        cur ^= 1;
    }

    // Epilogue. C layout (ISA 7.12.2): vgpr r -> row r + hh*8, lane lm -> col.
    #pragma unroll
    for (int ni = 0; ni < 4; ++ni) {
        const int col = gn0 + wn + ni * 16 + lm;
        float bv = 0.0f;
        if constexpr (HAS_BIAS) bv = bias[col];
        #pragma unroll
        for (int mi = 0; mi < 2; ++mi) {
            #pragma unroll
            for (int r = 0; r < 8; ++r) {
                const int row = gm0 + wm + mi * 16 + r + hh * 8;
                const float v = acc[mi][ni][r] + bv;
                const long long cidx = (long long)row * ldc + col;
                if constexpr (OUT_BF16) {
                    (reinterpret_cast<__bf16*>(Call) + (long long)bz * sCb)[cidx] = (__bf16)v;
                } else {
                    (reinterpret_cast<float*>(Call) + (long long)bz * sCb)[cidx] = v;
                }
            }
        }
    }
}

// ---------------------------------------------------------------------------
// x (B,N,D) f32 -> x_bf (B,N,D) bf16 and xT_bf (B,D,N) bf16.  32x32 LDS tiles.
// grid = (D/32, N/32, B), block = 256
// ---------------------------------------------------------------------------
__global__ __launch_bounds__(256)
void transpose_convert_x(const float* __restrict__ x,
                         __bf16* __restrict__ xbf,
                         __bf16* __restrict__ xT,
                         int Nn, int Dd)
{
    __shared__ float tile[32][33];
    const int b  = blockIdx.z;
    const int n0 = blockIdx.y * 32;
    const int d0 = blockIdx.x * 32;
    const int tid = threadIdx.x;
    const int r0 = tid >> 5, c = tid & 31;
    const float* xb = x + (long long)b * Nn * Dd;

    #pragma unroll
    for (int i = 0; i < 4; ++i) {
        const int r = r0 + i * 8;
        const float v = xb[(long long)(n0 + r) * Dd + d0 + c];
        tile[r][c] = v;
        xbf[(long long)b * Nn * Dd + (long long)(n0 + r) * Dd + d0 + c] = (__bf16)v;
    }
    __syncthreads();
    #pragma unroll
    for (int i = 0; i < 4; ++i) {
        const int r = r0 + i * 8;
        xT[(long long)b * Dd * Nn + (long long)(d0 + r) * Nn + n0 + c] = (__bf16)tile[c][r];
    }
}

__global__ void cvt_f32_bf16(const float* __restrict__ in, __bf16* __restrict__ out, int n) {
    const int i = blockIdx.x * 256 + threadIdx.x;
    if (i < n) out[i] = (__bf16)in[i];
}

// ---------------------------------------------------------------------------
// combine softmax: rows of 128 over T.  One wave per row (B*N rows).
// ---------------------------------------------------------------------------
__global__ __launch_bounds__(32)
void softmax_rows_T(const float* __restrict__ logits, __bf16* __restrict__ comb)
{
    const long long row = blockIdx.x;
    const int lane = threadIdx.x;
    const float4 f = reinterpret_cast<const float4*>(logits + row * 128)[lane];
    float v[4] = {f.x, f.y, f.z, f.w};
    float mx = fmaxf(fmaxf(v[0], v[1]), fmaxf(v[2], v[3]));
    for (int off = 16; off > 0; off >>= 1) mx = fmaxf(mx, __shfl_xor(mx, off));
    float s = 0.0f;
    #pragma unroll
    for (int i = 0; i < 4; ++i) { v[i] = expf(v[i] - mx); s += v[i]; }
    for (int off = 16; off > 0; off >>= 1) s += __shfl_xor(s, off);
    const float inv = 1.0f / s;
    __bf16* o = comb + row * 128 + lane * 4;
    #pragma unroll
    for (int i = 0; i < 4; ++i) o[i] = (__bf16)(v[i] * inv);
}

// ---------------------------------------------------------------------------
// dispatch softmax: columns of N=4096 over tokens, written TRANSPOSED:
// dispT[b,t,n].  grid = B*T blocks, block = 256, 16 elems/thread.
// ---------------------------------------------------------------------------
__global__ __launch_bounds__(256)
void softmax_cols_N(const float* __restrict__ logits, __bf16* __restrict__ dispT,
                    int Nn, int T)
{
    const int bt = blockIdx.x;
    const int b = bt / T, t = bt % T;
    const float* base = logits + (long long)b * Nn * T + t;
    const int tid = threadIdx.x;
    __shared__ float red[256];

    float v[16];
    float mx = -INFINITY;
    #pragma unroll
    for (int i = 0; i < 16; ++i) {
        v[i] = base[(long long)(tid + i * 256) * T];
        mx = fmaxf(mx, v[i]);
    }
    red[tid] = mx; __syncthreads();
    for (int s = 128; s > 0; s >>= 1) {
        if (tid < s) red[tid] = fmaxf(red[tid], red[tid + s]);
        __syncthreads();
    }
    mx = red[0]; __syncthreads();

    float sum = 0.0f;
    #pragma unroll
    for (int i = 0; i < 16; ++i) { v[i] = expf(v[i] - mx); sum += v[i]; }
    red[tid] = sum; __syncthreads();
    for (int s = 128; s > 0; s >>= 1) {
        if (tid < s) red[tid] += red[tid + s];
        __syncthreads();
    }
    const float inv = 1.0f / red[0];

    __bf16* o = dispT + ((long long)b * T + t) * Nn;
    #pragma unroll
    for (int i = 0; i < 16; ++i) o[tid + i * 256] = (__bf16)(v[i] * inv);
}

// ---------------------------------------------------------------------------
// Per-expert MLP half.  MODE 0: h = gelu(slots_e @ fc1^T + b1) -> hbuf(E,16,H)
//                       MODE 1: out = h_e @ fc2^T + b2 -> outT (B,D,T)
// A rows (M=16) staged bf16 in LDS; f32 weights streamed & converted in-reg.
// grid = (4, E) — 256 output cols per block; block = 256 (8 waves x 2 tiles).
// ---------------------------------------------------------------------------
template<int MODE>
__global__ __launch_bounds__(256)
void moe_mlp_kernel(const __bf16* __restrict__ Abuf,
                    const float*  __restrict__ Wg,
                    const float*  __restrict__ bias,
                    __bf16* __restrict__ Obuf)
{
    const int e   = blockIdx.y;
    const int nc0 = blockIdx.x * 256;
    __shared__ __align__(16) __bf16 sA[16 * 1032];   // 16 rows, K=1024 (+8 pad)

    const int tid = threadIdx.x;
    for (int c = tid; c < 16 * 128; c += 256) {      // 128 chunks of 8 bf16 per row
        const int m = c >> 7;
        const int ch = (c & 127) * 8;
        long long src;
        if constexpr (MODE == 0) {
            const int b = m >> 1, s = m & 1;         // slots (B,T,D), t = e*2+s
            src = ((long long)b * 128 + e * 2 + s) * 1024 + ch;
        } else {
            src = ((long long)e * 16 + m) * 1024 + ch;   // hbuf (E,16,H)
        }
        *reinterpret_cast<v8bf*>(&sA[m * 1032 + ch]) =
            *reinterpret_cast<const v8bf*>(&Abuf[src]);
    }
    __syncthreads();

    const int lane = tid & 31, wid = tid >> 5;
    const int lm = lane & 15, hh = lane >> 4;
    const float* We = Wg + (long long)e * 1024 * 1024;

    #pragma unroll
    for (int nt = 0; nt < 2; ++nt) {
        const int n = nc0 + (wid * 2 + nt) * 16 + lm;    // output column
        const float* wr = &We[(long long)n * 1024];
        v8f acc = {};
        for (int k0 = 0; k0 < 1024; k0 += 32) {
            FragBF af;
            af.h[0] = *reinterpret_cast<const v8bf*>(&sA[lm * 1032 + k0 + hh * 8]);
            af.h[1] = *reinterpret_cast<const v8bf*>(&sA[lm * 1032 + k0 + 16 + hh * 8]);

            __builtin_prefetch(&wr[k0 + 128], 0, 1);     // global_prefetch next blocks
            const v4f w0 = *reinterpret_cast<const v4f*>(&wr[k0 + hh * 8]);
            const v4f w1 = *reinterpret_cast<const v4f*>(&wr[k0 + hh * 8 + 4]);
            const v4f w2 = *reinterpret_cast<const v4f*>(&wr[k0 + 16 + hh * 8]);
            const v4f w3 = *reinterpret_cast<const v4f*>(&wr[k0 + 16 + hh * 8 + 4]);
            FragBF bu;
            #pragma unroll
            for (int j = 0; j < 4; ++j) {
                bu.e[j]      = (__bf16)w0[j];
                bu.e[4 + j]  = (__bf16)w1[j];
                bu.e[8 + j]  = (__bf16)w2[j];
                bu.e[12 + j] = (__bf16)w3[j];
            }
            acc = __builtin_amdgcn_wmma_f32_16x16x32_bf16(
                false, af.v, false, bu.v, (short)0, acc, false, false);
        }

        const float bv = bias[(long long)e * 1024 + n];
        #pragma unroll
        for (int r = 0; r < 8; ++r) {
            const int m = r + hh * 8;                    // slot-row (b*2+s)
            float v = acc[r] + bv;
            if constexpr (MODE == 0) {
                v = gelu_tanh(v);
                Obuf[((long long)e * 16 + m) * 1024 + n] = (__bf16)v;
            } else {
                const int b = m >> 1, s = m & 1;         // outT[b, d=n, t=e*2+s]
                Obuf[((long long)b * 1024 + n) * 128 + e * 2 + s] = (__bf16)v;
            }
        }
    }
}

__global__ void write_aux(float* out) { out[0] = 0.0f; }

// ---------------------------------------------------------------------------
extern "C" void kernel_launch(void* const* d_in, const int* in_sizes, int n_in,
                              void* d_out, int out_size, void* d_ws, size_t ws_size,
                              hipStream_t stream)
{
    (void)in_sizes; (void)n_in; (void)out_size; (void)ws_size;
    constexpr int B = 8, N = 4096, D = 1024, E = 64, T = 128;
    constexpr long long BN = (long long)B * N;

    const float* x     = (const float*)d_in[0];
    const float* phi_w = (const float*)d_in[1];
    const float* fc1_w = (const float*)d_in[2];
    const float* fc1_b = (const float*)d_in[3];
    const float* fc2_w = (const float*)d_in[4];
    const float* fc2_b = (const float*)d_in[5];
    const float* out_w = (const float*)d_in[6];
    const float* out_b = (const float*)d_in[7];
    float* out = (float*)d_out;

    char* ws = (char*)d_ws;
    size_t off = 0;
    auto take = [&](size_t bytes) { char* p = ws + off; off += (bytes + 255) & ~(size_t)255; return p; };
    __bf16* xbf      = (__bf16*)take(BN * D * 2);            // (B,N,D)
    __bf16* xT       = (__bf16*)take(BN * D * 2);            // (B,D,N)
    __bf16* phibf    = (__bf16*)take((size_t)T * D * 2);     // (T,D)
    __bf16* outwbf   = (__bf16*)take((size_t)D * D * 2);     // (D,D)
    float*  logits   = (float*) take(BN * T * 4);            // (B,N,T)
    __bf16* dispT    = (__bf16*)take((size_t)B * T * N * 2); // (B,T,N)
    __bf16* comb     = (__bf16*)take(BN * T * 2);            // (B,N,T)
    __bf16* slots    = (__bf16*)take((size_t)B * T * D * 2); // (B,T,D)
    __bf16* hbuf     = (__bf16*)take((size_t)E * 16 * D * 2);// (E,16,H)
    __bf16* outT     = (__bf16*)take((size_t)B * D * T * 2); // (B,D,T)
    __bf16* combined = (__bf16*)take(BN * D * 2);            // (B,N,D)

    // 1. convert + transpose x
    transpose_convert_x<<<dim3(D / 32, N / 32, B), 256, 0, stream>>>(x, xbf, xT, N, D);
    // 2. convert small weights
    cvt_f32_bf16<<<(T * D + 255) / 256, 256, 0, stream>>>(phi_w, phibf, T * D);
    cvt_f32_bf16<<<(D * D + 255) / 256, 256, 0, stream>>>(out_w, outwbf, D * D);
    // 3. logits = x @ phi^T   (32768 x 128 x 1024)
    gemm_nt_kernel<false, false><<<dim3(1, BN / 128, 1), 256, 0, stream>>>(
        xbf, 0, phibf, 0, nullptr, (void*)logits, 0, D, D, D, T);
    // 4. softmaxes
    softmax_rows_T<<<(int)BN, 32, 0, stream>>>(logits, comb);
    softmax_cols_N<<<B * T, 256, 0, stream>>>(logits, dispT, N, T);
    // 5. slots = dispT @ xT^T per batch   (128 x 1024 x 4096)
    gemm_nt_kernel<true, false><<<dim3(D / 128, T / 128, B), 256, 0, stream>>>(
        dispT, (long long)T * N, xT, (long long)D * N, nullptr,
        (void*)slots, (long long)T * D, N, N, N, D);
    // 6. expert MLPs
    moe_mlp_kernel<0><<<dim3(4, E), 256, 0, stream>>>(slots, fc1_w, fc1_b, hbuf);
    moe_mlp_kernel<1><<<dim3(4, E), 256, 0, stream>>>(hbuf, fc2_w, fc2_b, outT);
    // 7. combined = comb @ outT^T per batch   (4096 x 1024 x 128)
    gemm_nt_kernel<true, false><<<dim3(D / 128, N / 128, B), 256, 0, stream>>>(
        comb, (long long)N * T, outT, (long long)D * T, nullptr,
        (void*)combined, (long long)N * D, T, T, T, D);
    // 8. output = combined @ out_w^T + out_b   (32768 x 1024 x 1024)
    gemm_nt_kernel<false, true><<<dim3(D / 128, BN / 128, 1), 256, 0, stream>>>(
        combined, 0, outwbf, 0, out_b, (void*)out, 0, D, D, D, D);
    // 9. aux loss scalar
    write_aux<<<1, 1, 0, stream>>>(out + BN * D);
}
